// Attention_49185965474367
// MI455X (gfx1250) — compile-verified, matrix operand order checked by reading, not run on hardware
//

#include <hip/hip_runtime.h>
#include <hip/hip_bf16.h>
#include <math.h>

// ---------------------------------------------------------------------------
// Problem constants (match reference)
// ---------------------------------------------------------------------------
#define BATCH 4
#define SEQ   1024
#define DIM   1024
#define HEADS 16
#define DHEAD 64
#define INNER (HEADS * DHEAD)     // 1024
#define MROWS (BATCH * SEQ)       // 4096
#define SCALE 0.125f              // DHEAD^-0.5
#define NEGMAX (-3.402823466e+38f)

typedef __attribute__((ext_vector_type(16))) _Float16 v16h;
typedef __attribute__((ext_vector_type(8)))  _Float16 v8h;
typedef __attribute__((ext_vector_type(8)))  float    v8f;

#define WMMA_F16(a, b, c) \
    __builtin_amdgcn_wmma_f32_16x16x32_f16(false, (a), false, (b), (short)0, (c), false, false)

// ---------------------------------------------------------------------------
// Pre-pass 1: straight f32 -> f16 convert (x). 8 elements / thread, b128 i/o.
// ---------------------------------------------------------------------------
__global__ __launch_bounds__(256) void cvt_f32_f16_kernel(
    const float* __restrict__ in, _Float16* __restrict__ out)
{
    const int i = blockIdx.x * 256 + threadIdx.x;      // one v8h per thread
    const float4* p = (const float4*)in + (size_t)i * 2;
    const float4 a = p[0], b = p[1];
    v8h o;
    o[0] = (_Float16)a.x; o[1] = (_Float16)a.y;
    o[2] = (_Float16)a.z; o[3] = (_Float16)a.w;
    o[4] = (_Float16)b.x; o[5] = (_Float16)b.y;
    o[6] = (_Float16)b.z; o[7] = (_Float16)b.w;
    ((v8h*)out)[i] = o;
}

// ---------------------------------------------------------------------------
// Pre-pass 2: transpose + convert  out[c][r] = (f16) in[r][c].
// 32x32 LDS tile, 256 threads, coalesced both sides.
// ---------------------------------------------------------------------------
__global__ __launch_bounds__(256) void transpose_cvt_kernel(
    const float* __restrict__ in, _Float16* __restrict__ out, int R, int C)
{
    __shared__ float tile[32][33];
    const int tx = threadIdx.x & 31;
    const int ty = threadIdx.x >> 5;                   // 0..7
    const int c0 = blockIdx.x * 32;
    const int r0 = blockIdx.y * 32;
    #pragma unroll
    for (int k = 0; k < 4; ++k)
        tile[ty + 8 * k][tx] = in[(size_t)(r0 + ty + 8 * k) * C + c0 + tx];
    __syncthreads();
    #pragma unroll
    for (int k = 0; k < 4; ++k)
        out[(size_t)(c0 + ty + 8 * k) * R + r0 + tx] =
            (_Float16)tile[tx][ty + 8 * k];
}

// ---------------------------------------------------------------------------
// Fragment helpers: a lane's fragment = two contiguous v8h (16B) chunks.
// A/B 16-bit layout: lane m = lane&15, khalf = (lane>>4)*8;
//   halves 0-7  <-> K = khalf+i,  halves 8-15 <-> K = 16+khalf+i.
// ---------------------------------------------------------------------------
__device__ __forceinline__ v16h load_frag(const _Float16* p /*16B aligned*/)
{
    const v8h* v = (const v8h*)p;
    const v8h lo = v[0], hi = v[2];      // +0 and +16 halves
    v16h f;
    #pragma unroll
    for (int i = 0; i < 8; ++i) { f[i] = lo[i]; f[8 + i] = hi[i]; }
    return f;
}

// ---------------------------------------------------------------------------
// Kernel 1: fused QKV projection, all-f16 operands.
//   C[4096, 3072] = xh[4096,1024] @ [Wq | Wkv]^T-stored
//   cols [0,1024)    -> Q stored [bh][n][64]
//   cols [1024,2048) -> K stored [bh][n][64]   (natural: QK^T B-frag = row of K)
//   cols [2048,3072) -> V stored [bh][64][n]   (transposed: PV B-frag contiguous)
// One wave per block; 32x64 tile, K-step 32 -> 8 WMMA / 12 b128 loads per step.
// ---------------------------------------------------------------------------
__global__ __launch_bounds__(32) void qkv_proj_kernel(
    const _Float16* __restrict__ xh, const _Float16* __restrict__ wqt,
    const _Float16* __restrict__ wkvt,
    _Float16* __restrict__ qbuf, _Float16* __restrict__ kbuf,
    _Float16* __restrict__ vtbuf)
{
    const int lane  = threadIdx.x & 31;
    const int m     = lane & 15;
    const int khalf = (lane >> 4) * 8;
    const int row0  = blockIdx.x * 32;
    const int c0    = blockIdx.y * 64;

    v8f acc[2][4] = {};

    for (int k0 = 0; k0 < DIM; k0 += 32) {
        const v16h a0 = load_frag(xh + (size_t)(row0 + m) * DIM + k0 + khalf);
        const v16h a1 = load_frag(xh + (size_t)(row0 + 16 + m) * DIM + k0 + khalf);
        #pragma unroll
        for (int t = 0; t < 4; ++t) {
            const int c = c0 + 16 * t + m;
            const _Float16* wrow = (c < INNER)
                ? (wqt + (size_t)c * DIM)
                : (wkvt + (size_t)(c - INNER) * DIM);
            const v16h bf = load_frag(wrow + k0 + khalf);
            acc[0][t] = WMMA_F16(a0, bf, acc[0][t]);
            acc[1][t] = WMMA_F16(a1, bf, acc[1][t]);
        }
    }

    #pragma unroll
    for (int half = 0; half < 2; ++half) {
        #pragma unroll
        for (int t = 0; t < 4; ++t) {
            const int c = c0 + 16 * t + m;
            #pragma unroll
            for (int r = 0; r < 8; ++r) {
                const int row = row0 + 16 * half + khalf + r;
                const int bb  = row >> 10;
                const int n   = row & (SEQ - 1);
                const _Float16 val = (_Float16)acc[half][t][r];
                if (c < INNER) {
                    const int h = c >> 6, dd = c & 63;
                    qbuf[((size_t)(bb * HEADS + h) * SEQ + n) * DHEAD + dd] = val;
                } else if (c < 2 * INNER) {
                    const int cc = c - INNER;
                    const int h = cc >> 6, dd = cc & 63;
                    kbuf[((size_t)(bb * HEADS + h) * SEQ + n) * DHEAD + dd] = val;
                } else {
                    const int cc = c - 2 * INNER;
                    const int h = cc >> 6, dd = cc & 63;
                    vtbuf[((size_t)(bb * HEADS + h) * DHEAD + dd) * SEQ + n] = val;
                }
            }
        }
    }
}

// ---------------------------------------------------------------------------
// Kernel 2: streaming (flash-style) attention per (bh, 16-row tile).
//   rel_pos (256 MB) streamed exactly once; scores never touch memory.
// All K/V fragments are contiguous-per-lane b128 loads.
// ---------------------------------------------------------------------------
__global__ __launch_bounds__(32) void attn_kernel(
    const _Float16* __restrict__ qbuf, const _Float16* __restrict__ kbuf,
    const _Float16* __restrict__ vtbuf, const float* __restrict__ rel_pos,
    const unsigned char* __restrict__ qmask,
    const unsigned char* __restrict__ cmask,
    _Float16* __restrict__ obuf)
{
    __shared__ _Float16 Pt[16][32];             // P relayout buffer (1 KB)

    const int lane  = threadIdx.x & 31;
    const int m     = lane & 15;
    const int khalf = (lane >> 4) * 8;
    const int i0    = blockIdx.x * 16;
    const int bh    = blockIdx.y;
    const int b     = bh >> 4;

    // Q fragments, K-dim 0..31 / 32..63
    const _Float16* qrow = qbuf + ((size_t)bh * SEQ + i0 + m) * DHEAD;
    const v16h aq0 = load_frag(qrow + khalf);
    const v16h aq1 = load_frag(qrow + 32 + khalf);

    bool qm[8];
    #pragma unroll
    for (int r = 0; r < 8; ++r)
        qm[r] = qmask[b * SEQ + i0 + khalf + r] != 0;

    v8f   oacc[4] = {};
    float mst[8], lst[8];
    #pragma unroll
    for (int r = 0; r < 8; ++r) { mst[r] = -INFINITY; lst[r] = 0.0f; }

    const _Float16* kb  = kbuf  + (size_t)bh * SEQ * DHEAD;
    const _Float16* vt  = vtbuf + (size_t)bh * DHEAD * SEQ;
    const float*    rpb = rel_pos + ((size_t)bh * SEQ + i0 + khalf) * SEQ;

    for (int j0 = 0; j0 < SEQ; j0 += 32) {
        // ---- B frags of K^T: column j of K^T == row j of K (contiguous) ----
        const _Float16* krA = kb + (size_t)(j0 + m) * DHEAD;        // tile j0..+15
        const _Float16* krB = kb + (size_t)(j0 + 16 + m) * DHEAD;   // tile +16..+31
        const v16h bka0 = load_frag(krA + khalf);
        const v16h bka1 = load_frag(krA + 32 + khalf);
        const v16h bkb0 = load_frag(krB + khalf);
        const v16h bkb1 = load_frag(krB + 32 + khalf);

        v8f sa = {}, sb = {};
        sa = WMMA_F16(aq0, bka0, sa);
        sa = WMMA_F16(aq1, bka1, sa);
        sb = WMMA_F16(aq0, bkb0, sb);
        sb = WMMA_F16(aq1, bkb1, sb);

        // ---- bias + masks; rel_pos streamed once, prefetch next block ----
        const bool cma = cmask[b * SEQ + j0 + m] != 0;
        const bool cmb = cmask[b * SEQ + j0 + 16 + m] != 0;
        if (j0 + 32 < SEQ)
            __builtin_prefetch(rpb + j0 + 32 + m, 0, 1);

        float saf[8], sbf[8];
        #pragma unroll
        for (int r = 0; r < 8; ++r) {
            const float* rp = rpb + (size_t)r * SEQ + j0;
            const float va = sa[r] * SCALE + rp[m];
            const float vb = sb[r] * SCALE + rp[16 + m];
            saf[r] = (qm[r] && cma) ? va : NEGMAX;
            sbf[r] = (qm[r] && cmb) ? vb : NEGMAX;
        }

        // ---- online softmax (row = 16-lane half-wave; xor-shuffle reduce) ----
        #pragma unroll
        for (int r = 0; r < 8; ++r) {
            float mx = fmaxf(saf[r], sbf[r]);
            #pragma unroll
            for (int off = 8; off >= 1; off >>= 1)
                mx = fmaxf(mx, __shfl_xor(mx, off, 32));
            const float newm  = fmaxf(mst[r], mx);
            const float alpha = __expf(mst[r] - newm);
            const float pa = __expf(saf[r] - newm);
            const float pb = __expf(sbf[r] - newm);
            float rs = pa + pb;
            #pragma unroll
            for (int off = 8; off >= 1; off >>= 1)
                rs += __shfl_xor(rs, off, 32);
            lst[r] = lst[r] * alpha + rs;
            mst[r] = newm;
            #pragma unroll
            for (int t = 0; t < 4; ++t) oacc[t][r] *= alpha;
            Pt[khalf + r][m]      = (_Float16)pa;
            Pt[khalf + r][16 + m] = (_Float16)pb;
        }
        __syncthreads();

        // ---- P back in A-fragment layout (two ds_load_b128) ----
        const v16h ap = load_frag(&Pt[m][khalf]);
        __syncthreads();

        // ---- O += P @ V_block : V^T rows are contiguous per lane ----
        #pragma unroll
        for (int t = 0; t < 4; ++t) {
            const int dd = 16 * t + m;
            const v16h bv = load_frag(vt + (size_t)dd * SEQ + j0 + khalf);
            oacc[t] = WMMA_F16(ap, bv, oacc[t]);
        }
    }

    // ---- epilogue: normalized O, merged-head layout [b][n][h*64+d] ----
    const int h = bh & 15;
    #pragma unroll
    for (int r = 0; r < 8; ++r) {
        const float inv = 1.0f / lst[r];
        _Float16* orow =
            obuf + ((size_t)(b * SEQ + i0 + khalf + r)) * INNER + h * DHEAD;
        #pragma unroll
        for (int t = 0; t < 4; ++t)
            orow[16 * t + m] = (_Float16)(oacc[t][r] * inv);
    }
}

// ---------------------------------------------------------------------------
// Kernel 3: output projection  Y = O(f16) @ Wo + bo, f32 out. 32x64 tile.
// ---------------------------------------------------------------------------
__global__ __launch_bounds__(32) void out_proj_kernel(
    const _Float16* __restrict__ obuf, const _Float16* __restrict__ wot,
    const float* __restrict__ bo, float* __restrict__ y)
{
    const int lane  = threadIdx.x & 31;
    const int m     = lane & 15;
    const int khalf = (lane >> 4) * 8;
    const int row0  = blockIdx.x * 32;
    const int c0    = blockIdx.y * 64;

    v8f acc[2][4] = {};

    for (int k0 = 0; k0 < INNER; k0 += 32) {
        const v16h a0 = load_frag(obuf + (size_t)(row0 + m) * INNER + k0 + khalf);
        const v16h a1 = load_frag(obuf + (size_t)(row0 + 16 + m) * INNER + k0 + khalf);
        #pragma unroll
        for (int t = 0; t < 4; ++t) {
            const int c = c0 + 16 * t + m;
            const v16h bf = load_frag(wot + (size_t)c * INNER + k0 + khalf);
            acc[0][t] = WMMA_F16(a0, bf, acc[0][t]);
            acc[1][t] = WMMA_F16(a1, bf, acc[1][t]);
        }
    }

    #pragma unroll
    for (int half = 0; half < 2; ++half) {
        #pragma unroll
        for (int t = 0; t < 4; ++t) {
            const int c = c0 + 16 * t + m;
            const float bias = bo[c];
            #pragma unroll
            for (int r = 0; r < 8; ++r)
                y[(size_t)(row0 + 16 * half + khalf + r) * DIM + c] =
                    acc[half][t][r] + bias;
        }
    }
}

// ---------------------------------------------------------------------------
// Launcher
// ---------------------------------------------------------------------------
extern "C" void kernel_launch(void* const* d_in, const int* in_sizes, int n_in,
                              void* d_out, int out_size, void* d_ws, size_t ws_size,
                              hipStream_t stream) {
    const float*         x       = (const float*)d_in[0];
    const float*         rel_pos = (const float*)d_in[1];
    const unsigned char* qmask   = (const unsigned char*)d_in[2];
    const unsigned char* cmask   = (const unsigned char*)d_in[3];
    const float*         Wq      = (const float*)d_in[4];
    const float*         Wkv     = (const float*)d_in[5];
    const float*         Wo      = (const float*)d_in[6];
    const float*         bo      = (const float*)d_in[7];
    float*               y       = (float*)d_out;

    const size_t MI = (size_t)MROWS * INNER;          // 4M elements
    _Float16* xh    = (_Float16*)d_ws;                //  8 MB
    _Float16* wqt   = xh    + MI;                     //  2 MB  [1024][1024]
    _Float16* wkvt  = wqt   + (size_t)INNER * DIM;    //  4 MB  [2048][1024]
    _Float16* wot   = wkvt  + (size_t)2 * INNER * DIM;//  2 MB  [1024][1024]
    _Float16* qbuf  = wot   + (size_t)DIM * INNER;    //  8 MB
    _Float16* kbuf  = qbuf  + MI;                     //  8 MB
    _Float16* vtbuf = kbuf  + MI;                     //  8 MB
    _Float16* obuf  = vtbuf + MI;                     //  8 MB  (48 MB total)

    // 0) precision pre-pass: x -> f16; weights -> f16, transposed [out][k]
    cvt_f32_f16_kernel<<<dim3(MROWS * DIM / (256 * 8)), 256, 0, stream>>>(x, xh);
    transpose_cvt_kernel<<<dim3(INNER / 32, DIM / 32), 256, 0, stream>>>(
        Wq, wqt, DIM, INNER);
    transpose_cvt_kernel<<<dim3(2 * INNER / 32, DIM / 32), 256, 0, stream>>>(
        Wkv, wkvt, DIM, 2 * INNER);
    transpose_cvt_kernel<<<dim3(DIM / 32, INNER / 32), 256, 0, stream>>>(
        Wo, wot, INNER, DIM);

    // 1) QKV projection: 128 row tiles x 48 col tiles
    qkv_proj_kernel<<<dim3(MROWS / 32, 3 * INNER / 64), 32, 0, stream>>>(
        xh, wqt, wkvt, qbuf, kbuf, vtbuf);

    // 2) streaming attention: 64 row tiles x 64 (b,h) pairs
    attn_kernel<<<dim3(SEQ / 16, BATCH * HEADS), 32, 0, stream>>>(
        qbuf, kbuf, vtbuf, rel_pos, qmask, cmask, obuf);

    // 3) output projection: 128 row tiles x 16 col tiles
    out_proj_kernel<<<dim3(MROWS / 32, DIM / 64), 32, 0, stream>>>(
        obuf, wot, bo, y);
}
